// CompressedInteractionNetwork_89988154786030
// MI455X (gfx1250) — compile-verified, matrix operand order checked by reading, not run on hardware
//
#include <hip/hip_runtime.h>

// ---------------------------------------------------------------------------
// CIN (xDeepFM) forward for B=2048, F0=39, D=16, layers [128,128,128].
// Per-batch GEMM  Z(128x16) = W^T(128xHK) x O(HKx16),  O[(h,k),d]=x0[h,d]*xk[k,d]
// bf16 WMMA (v_wmma_f32_16x16x32_bf16) with fp32 accumulate.
// ---------------------------------------------------------------------------

typedef __attribute__((ext_vector_type(16))) __bf16          v16bf;
typedef __attribute__((ext_vector_type(8)))  float           v8f;
typedef __attribute__((ext_vector_type(16))) unsigned short  v16u;
typedef __attribute__((ext_vector_type(8)))  unsigned short  v8u;

#define NBATCH 2048
#define NF0    39
#define ND     16
#define NS     128
#define SOP    1256   // O row stride in ushorts (2512 B: 16B-aligned, bank-conflict-free)

__device__ __forceinline__ unsigned short f2bf(float f) {
  // round-to-nearest-even fp32 -> bf16 (matches WMMA RNE convention)
  unsigned u = __builtin_bit_cast(unsigned, f);
  u += 0x7FFFu + ((u >> 16) & 1u);
  return (unsigned short)(u >> 16);
}

// ---------------------------------------------------------------------------
// Prep: repack one layer's W (fp32 [h][k][s], s fastest) into bf16 A-fragments
// in the exact CDNA5 16-bit A-matrix (16x32) lane/VGPR layout, zero-padded in K.
// Fragment (kk, stile): 32 lanes x 16 halves, stored contiguously (1024 B).
// Grid: Ksteps blocks x 256 threads (8 stiles x 32 lanes per block).
// ---------------------------------------------------------------------------
__global__ void __launch_bounds__(256)
cin_prep(const float* __restrict__ w, unsigned short* __restrict__ wt,
         int HK, int Ksteps) {
  const int kk    = blockIdx.x;
  const int stile = threadIdx.x >> 5;
  const int lane  = threadIdx.x & 31;
  const int m     = lane & 15;       // A-matrix row (within 16-row tile)
  const int khalf = lane >> 4;
  const int s     = stile * 16 + m;  // output-channel index

  v16u frag;
#pragma unroll
  for (int j = 0; j < 8; ++j) {
    // K pair held by VGPR j for this lane (ISA 16-bit A layout)
    const int kb = ((j & 3) << 1) + (khalf << 3) + ((j >> 2) << 4);
    const int K0 = kk * 32 + kb;
#pragma unroll
    for (int t = 0; t < 2; ++t) {
      const int K = K0 + t;                     // flattened (h,k) = K
      const float v = (K < HK) ? w[K * NS + s] : 0.0f;
      frag[2 * j + t] = f2bf(v);
    }
  }
  unsigned short* dst = wt + (((size_t)(stile * Ksteps + kk) * 32 + lane) * 16);
  *(v16u*)dst = frag;   // 32 B per lane -> 2x global_store_b128, fully coalesced
}

// ---------------------------------------------------------------------------
// Main: one workgroup (8 waves) per batch. Three layers in sequence.
// ---------------------------------------------------------------------------
__global__ void __launch_bounds__(256)
cin_main(const float* __restrict__ inputs,
         const float* __restrict__ bias0, const float* __restrict__ bias1,
         const float* __restrict__ bias2,
         const unsigned short* __restrict__ wt0,
         const unsigned short* __restrict__ wt1,
         const unsigned short* __restrict__ wt2,
         float* __restrict__ out) {
  __shared__ float          s_x0[NF0 * ND];   // 2496 B
  __shared__ float          s_xk[64 * ND];    // 4096 B
  __shared__ unsigned short s_O[ND * SOP];    // 40192 B  (one K-chunk of O, bf16)

  const int b     = blockIdx.x;
  const int tid   = threadIdx.x;
  const int lane  = tid & 31;
  const int wave  = tid >> 5;      // = S-tile index (8 tiles of 16 rows = 128)
  const int dl    = lane & 15;     // C/D + B layout: lane%16 = N column = d
  const int khalf = lane >> 4;

  for (int i = tid; i < NF0 * ND; i += 256)
    s_x0[i] = inputs[(size_t)b * (NF0 * ND) + i];

  const int FkA[3]                  = {NF0, 64, 64};
  const int HKA[3]                  = {1521, 2496, 2496};
  const int CSA[3]                  = {24, 39, 39};   // K-steps per chunk (2 chunks)
  const unsigned short* wtA[3]      = {wt0, wt1, wt2};
  const float* biasA[3]             = {bias0, bias1, bias2};

  for (int l = 0; l < 3; ++l) {
    const int Fk      = FkA[l];
    const int HK      = HKA[l];
    const int csteps  = CSA[l];
    const int Ksteps  = 2 * csteps;
    const float* xk   = (l == 0) ? s_x0 : s_xk;
    const unsigned short* wt = wtA[l];

    v8f acc = {};

    for (int c = 0; c < 2; ++c) {
      __syncthreads();   // prior chunk's ds_loads / epilogue xk writes complete

      // ---- build O chunk in LDS: O[d][kloc] = bf16(x0[h,d]*xk[k,d]) --------
      const int kc0   = c * csteps * 32;
      const int pairs = (csteps * 32) >> 1;
      for (int wi = tid; wi < ND * pairs; wi += 256) {
        const int d  = wi & 15;
        const int p  = wi >> 4;
        const int h0 = kc0 + 2 * p;
        unsigned pack = 0;
#pragma unroll
        for (int t = 0; t < 2; ++t) {
          const int hk = h0 + t;
          float v = 0.0f;
          if (hk < HK) {
            int h, k;
            if (Fk == 64) { h = hk >> 6; k = hk & 63; }
            else { h = (int)(((unsigned)hk * 1681u) >> 16); k = hk - h * 39; } // exact for hk<1521
            v = s_x0[h * ND + d] * xk[k * ND + d];
          }
          pack |= ((unsigned)f2bf(v)) << (16 * t);
        }
        *(unsigned*)&s_O[d * SOP + 2 * p] = pack;
      }
      __syncthreads();

      // ---- WMMA over chunk: wave = S-tile, K loop ---------------------------
      const unsigned short* wtw =
          wt + (((size_t)(wave * Ksteps + c * csteps) * 32 + lane) * 16);
      for (int kk = 0; kk < csteps; ++kk) {
        // A fragment: pre-swizzled bf16 from global (L2/WGP$-resident)
        v16u av = *(const v16u*)(wtw + (size_t)kk * 512);
        __builtin_prefetch(wtw + (size_t)(kk + 1) * 512, 0, 1);

        // B fragment: ISA 16-bit B layout; K groups of 8 are contiguous here
        const unsigned short* ob = &s_O[dl * SOP + kk * 32 + khalf * 8];
        union { v8u h[2]; v16u v; } bu;
        bu.h[0] = *(const v8u*)ob;          // K = khalf*8 + 0..7   (VGPRs 0..3)
        bu.h[1] = *(const v8u*)(ob + 16);   // K = 16+khalf*8+0..7  (VGPRs 4..7)

        acc = __builtin_amdgcn_wmma_f32_16x16x32_bf16(
            false, __builtin_bit_cast(v16bf, av),
            false, __builtin_bit_cast(v16bf, bu.v),
            (short)0, acc, false, false);
      }
    }

    // ---- epilogue: bias + ReLU, split xk/direct, sum over d ----------------
    const float* bias = biasA[l];
#pragma unroll
    for (int j = 0; j < 8; ++j) {
      const int s = wave * 16 + j + 8 * khalf;   // C/D layout: VGPR j row
      float v = acc[j] + bias[s];
      v = v > 0.0f ? v : 0.0f;
      if (l < 2 && s < 64) s_xk[s * ND + dl] = v;  // feed next layer (fp32)

      float r = v;                                  // Σ over d (16 lanes/group)
      r += __shfl_xor(r, 1, 32);
      r += __shfl_xor(r, 2, 32);
      r += __shfl_xor(r, 4, 32);
      r += __shfl_xor(r, 8, 32);
      if (dl == 0) {
        if (l < 2) {
          if (s >= 64) out[(size_t)b * 256 + l * 64 + (s - 64)] = r;
        } else {
          out[(size_t)b * 256 + 128 + s] = r;
        }
      }
    }
  }
}

// ---------------------------------------------------------------------------
// Launch
// ---------------------------------------------------------------------------
extern "C" void kernel_launch(void* const* d_in, const int* in_sizes, int n_in,
                              void* d_out, int out_size, void* d_ws, size_t ws_size,
                              hipStream_t stream) {
  const float* inputs = (const float*)d_in[0];
  const float* w0 = (const float*)d_in[1];
  const float* b0 = (const float*)d_in[2];
  const float* w1 = (const float*)d_in[3];
  const float* b1 = (const float*)d_in[4];
  const float* w2 = (const float*)d_in[5];
  const float* b2 = (const float*)d_in[6];
  float* out = (float*)d_out;

  // Packed bf16 A-fragment buffers in workspace (~1.63 MB total)
  unsigned short* wt0 = (unsigned short*)d_ws;
  unsigned short* wt1 = wt0 + (size_t)8 * 48 * 32 * 16;   // 393216 B
  unsigned short* wt2 = wt1 + (size_t)8 * 78 * 32 * 16;   // 638976 B each

  cin_prep<<<48, 256, 0, stream>>>(w0, wt0, 1521, 48);
  cin_prep<<<78, 256, 0, stream>>>(w1, wt1, 2496, 78);
  cin_prep<<<78, 256, 0, stream>>>(w2, wt2, 2496, 78);

  cin_main<<<NBATCH, 256, 0, stream>>>(inputs, b0, b1, b2, wt0, wt1, wt2, out);
}
